// BinarizeLinear_inference_5171140625022
// MI455X (gfx1250) — 8-bit matrix operand order repaired and checked by reading, compiled, not run on hardware
//
#include <hip/hip_runtime.h>
#include <stdint.h>

typedef __attribute__((ext_vector_type(16))) int   v16i;
typedef __attribute__((ext_vector_type(8)))  float v8f;

static constexpr int Mdim = 8192;
static constexpr int Kdim = 4096;
static constexpr int Ndim = 4096;

static constexpr int TM = 128;
static constexpr int TN = 64;
static constexpr int TK = 128;
static constexpr int LDSS   = 144;
static constexpr int A_TILE = TM * LDSS;
static constexpr int B_TILE = TN * LDSS;
static constexpr int BUFB   = A_TILE + B_TILE;

#define FP8_P1 0x38u
#define FP8_M1 0xB8u

__global__ void binarize_kernel(const float* __restrict__ in,
                                unsigned char* __restrict__ out, int n4) {
  int i = blockIdx.x * blockDim.x + threadIdx.x;
  if (i < n4) {
    float4 v = ((const float4*)in)[i];
    unsigned b0 = v.x > 0.f ? FP8_P1 : FP8_M1;
    unsigned b1 = v.y > 0.f ? FP8_P1 : FP8_M1;
    unsigned b2 = v.z > 0.f ? FP8_P1 : FP8_M1;
    unsigned b3 = v.w > 0.f ? FP8_P1 : FP8_M1;
    ((unsigned*)out)[i] = b0 | (b1 << 8) | (b2 << 16) | (b3 << 24);
  }
}

__global__ void binarize_transpose_kernel(const float* __restrict__ w,
                                          unsigned char* __restrict__ wt) {
  __shared__ unsigned char tile[64][68];
  const int n0 = blockIdx.x * 64;
  const int k0 = blockIdx.y * 64;
  const int tx = threadIdx.x & 63;
  const int ty = threadIdx.x >> 6;
#pragma unroll
  for (int r = 0; r < 16; ++r) {
    int k = ty * 16 + r;
    float v = w[(size_t)(k0 + k) * Ndim + (n0 + tx)];
    tile[k][tx] = v > 0.f ? (unsigned char)FP8_P1 : (unsigned char)FP8_M1;
  }
  __syncthreads();
#pragma unroll
  for (int r = 0; r < 16; ++r) {
    int n = ty * 16 + r;
    wt[(size_t)(n0 + n) * Kdim + (k0 + tx)] = tile[tx][n];
  }
}

__global__ __launch_bounds__(256)
void bgemm_fp8_wmma(const unsigned char* __restrict__ A,
                    const unsigned char* __restrict__ Bt,
                    const float* __restrict__ bias,
                    float* __restrict__ C) {
  __shared__ unsigned char smem[2 * BUFB];
  const int tid  = threadIdx.x;
  const int lane = tid & 31;
  const int l16  = lane & 15;
  const int lhi  = lane >> 4;
  const int wave = tid >> 5;
  const int wm   = wave >> 1;
  const int wn   = wave & 1;

  const unsigned char* Ag = A  + (size_t)blockIdx.y * TM * Kdim;
  const unsigned char* Bg = Bt + (size_t)blockIdx.x * TN * Kdim;

  const unsigned lds_base = (unsigned)(uintptr_t)&smem[0];

  const unsigned char* gA[4];
  const unsigned char* gB[2];
  unsigned offA[4], offB[2];
#pragma unroll
  for (int i = 0; i < 4; ++i) {
    int c = tid + 256 * i, r = c >> 3, cc = (c & 7) * 16;
    gA[i]   = Ag + (size_t)r * Kdim + cc;
    offA[i] = lds_base + (unsigned)(r * LDSS + cc);
  }
#pragma unroll
  for (int i = 0; i < 2; ++i) {
    int c = tid + 256 * i, r = c >> 3, cc = (c & 7) * 16;
    gB[i]   = Bg + (size_t)r * Kdim + cc;
    offB[i] = lds_base + (unsigned)(A_TILE + r * LDSS + cc);
  }

  auto issue_tile = [&](int buf, int kt) {
    const size_t   koff = (size_t)kt * TK;
    const unsigned lb   = (unsigned)(buf * BUFB);
#pragma unroll
    for (int i = 0; i < 4; ++i) {
      asm volatile("global_load_async_to_lds_b128 %0, %1, off"
                   :: "v"(offA[i] + lb), "v"(gA[i] + koff) : "memory");
    }
#pragma unroll
    for (int i = 0; i < 2; ++i) {
      asm volatile("global_load_async_to_lds_b128 %0, %1, off"
                   :: "v"(offB[i] + lb), "v"(gB[i] + koff) : "memory");
    }
  };

  v8f acc[2][2] = {};

  auto compute = [&](int buf) {
    const unsigned char* As = &smem[buf * BUFB];
    const unsigned char* Bs = As + A_TILE;
    v16i bf[2];
#pragma unroll
    for (int n = 0; n < 2; ++n) {
      const unsigned char* row = Bs + (wn * 32 + n * 16 + l16) * LDSS + lhi * 16;
      union { v16i v; uint4 q[4]; } u;
#pragma unroll
      for (int q = 0; q < 4; ++q) u.q[q] = *(const uint4*)(row + q * 32);
      bf[n] = u.v;
    }
#pragma unroll
    for (int m = 0; m < 2; ++m) {
      const unsigned char* row = As + (wm * 32 + m * 16 + l16) * LDSS + lhi * 16;
      union { v16i v; uint2 d[8]; } u;
#pragma unroll
      for (int p = 0; p < 8; ++p) u.d[p] = *(const uint2*)(row + (p >> 1) * 32 + (p & 1) * 8);
#pragma unroll
      for (int n = 0; n < 2; ++n)
        acc[m][n] = __builtin_amdgcn_wmma_f32_16x16x128_fp8_fp8(
            u.v, bf[n], (short)0, acc[m][n], false, false);
    }
  };

  constexpr int KT = Kdim / TK;
  issue_tile(0, 0);
  int kt = 0;
  for (; kt < KT - 1; ++kt) {
    issue_tile((kt + 1) & 1, kt + 1);
    asm volatile("s_wait_asynccnt 0x6" ::: "memory");
    __syncthreads();
    compute(kt & 1);
    __syncthreads();
  }
  asm volatile("s_wait_asynccnt 0x0" ::: "memory");
  __syncthreads();
  compute(kt & 1);

  const int col0 = blockIdx.x * TN + wn * 32 + l16;
  const int row0 = blockIdx.y * TM + wm * 32 + lhi * 8;
#pragma unroll
  for (int n = 0; n < 2; ++n) {
    float bv = bias[col0 + n * 16];
#pragma unroll
    for (int m = 0; m < 2; ++m) {
#pragma unroll
      for (int j = 0; j < 8; ++j) {
        C[(size_t)(row0 + m * 16 + j) * Ndim + (col0 + n * 16)] = acc[m][n][j] + bv;
      }
    }
  }
}

extern "C" void kernel_launch(void* const* d_in, const int* in_sizes, int n_in,
                              void* d_out, int out_size, void* d_ws, size_t ws_size,
                              hipStream_t stream) {
  const float* input  = (const float*)d_in[0];
  const float* weight = (const float*)d_in[1];
  const float* bias   = (const float*)d_in[2];
  float* out = (float*)d_out;

  unsigned char* Ab = (unsigned char*)d_ws;
  unsigned char* Wt = Ab + (size_t)Mdim * Kdim;

  int n4 = (Mdim * Kdim) / 4;
  binarize_kernel<<<(n4 + 255) / 256, 256, 0, stream>>>(input, Ab, n4);

  dim3 tg(Ndim / 64, Kdim / 64);
  binarize_transpose_kernel<<<tg, 256, 0, stream>>>(weight, Wt);

  dim3 gg(Ndim / TN, Mdim / TM);
  bgemm_fp8_wmma<<<gg, 256, 0, stream>>>(Ab, Wt, bias, out);
}
